// MemTransformerLM_65566970741121
// MI455X (gfx1250) — compile-verified, matrix operand order checked by reading, not run on hardware
//
#include <hip/hip_runtime.h>

typedef unsigned short u16;
typedef unsigned int   u32;

typedef __attribute__((ext_vector_type(16))) __bf16 v16bf;
typedef __attribute__((ext_vector_type(16))) short  v16s;
typedef __attribute__((ext_vector_type(8)))  float  v8f;

#define QLEN 1024
#define MLEN 1024
#define BSZ  2
#define DMODEL 1024
#define NHEAD 16
#define DHEAD 64
#define DINNER 4096
#define KLEN 2048

__device__ __forceinline__ u16 f2bf(float f) {
    u32 u = __float_as_uint(f);
    u32 r = u + 0x7FFFu + ((u >> 16) & 1u);
    return (u16)(r >> 16);
}

__device__ __forceinline__ v8f wmma_bf16(v16s a, v16s b, v8f c) {
    return __builtin_amdgcn_wmma_f32_16x16x32_bf16(
        false, __builtin_bit_cast(v16bf, a),
        false, __builtin_bit_cast(v16bf, b),
        (short)0, c, false, false);
}

// Load a 16x32 bf16 fragment whose per-lane data lives at `base + sel*stride`
// (sel = lane%16 selects the M/N line, elements are contiguous in K).
// Per ISA 7.12.2 a lane holds K {0..7,16..23} (lanes 0-15) or {8..15,24..31}
// (lanes 16-31): two contiguous 16B runs -> two ds_load_b128.
__device__ __forceinline__ v16s frag_ld(const u16* base, int stride, int lane) {
    int sel = lane & 15;
    int khi = (lane & 16) ? 8 : 0;
    const u16* p = base + sel * stride + khi;
    uint4 lo = *(const uint4*)(p);        // K khi .. khi+7
    uint4 hi = *(const uint4*)(p + 16);   // K khi+16 .. khi+23
    struct { uint4 a, b; } pair{lo, hi};
    return __builtin_bit_cast(v16s, pair);
}

// ---------------------------------------------------------------- cvt f32->bf16
__global__ void cvt_bf16_kernel(const float* __restrict__ src, u16* __restrict__ dst, int n) {
    int i = blockIdx.x * blockDim.x + threadIdx.x;
    if (i < n) dst[i] = f2bf(src[i]);
}

// ---------------------------------------------------------------- generic WMMA GEMM
// C[M,N] = A[M,K](bf16) * B[K,N](bf16) + bias ; optional relu ; f32 and/or bf16 out
#define BM 128
#define BN 128
#define BK 32

__global__ __launch_bounds__(256) void gemm_bf16_kernel(
    const u16* __restrict__ A, int lda,
    const u16* __restrict__ B, int ldb,
    const float* __restrict__ bias,
    float* __restrict__ Cf, u16* __restrict__ Cb, int ldc,
    int M, int N, int K, int relu)
{
    __shared__ u16 As[BM][BK];     // row-major: [m][k]
    __shared__ u16 BsT[BN][BK];    // transposed: [n][k]

    int m0 = blockIdx.y * BM, n0 = blockIdx.x * BN;
    int t = threadIdx.x, lane = t & 31, wv = t >> 5;
    int wm = (wv & 3) * 32;   // 2 m-tiles per wave
    int wn = (wv >> 2) * 64;  // 4 n-tiles per wave
    int mloc = lane & 15;
    int mhi  = (lane & 16) ? 8 : 0;

    v8f acc[2][4];
#pragma unroll
    for (int i = 0; i < 2; i++)
#pragma unroll
        for (int j = 0; j < 4; j++)
#pragma unroll
            for (int g = 0; g < 8; g++) acc[i][j][g] = 0.0f;

    int arow = t >> 1, ac0 = (t & 1) * 16;
    int brow = t >> 3, bc0 = (t & 7) * 16;

    for (int k0 = 0; k0 < K; k0 += BK) {
        __syncthreads();
        {
            const uint4* s = (const uint4*)(A + (size_t)(m0 + arow) * lda + k0 + ac0);
            uint4* d = (uint4*)&As[arow][ac0];
            d[0] = s[0]; d[1] = s[1];
        }
        {
            // read 16 contiguous cols of B row (k0+brow), store transposed
            const u16* s = B + (size_t)(k0 + brow) * ldb + n0 + bc0;
            uint4 r0 = ((const uint4*)s)[0];
            uint4 r1 = ((const uint4*)s)[1];
            u16 tmp[16];
            *(uint4*)&tmp[0] = r0; *(uint4*)&tmp[8] = r1;
#pragma unroll
            for (int i = 0; i < 16; i++) BsT[bc0 + i][brow] = tmp[i];
        }
        if (k0 + BK < K) { // prefetch next k-tiles (global_prefetch_b8)
            __builtin_prefetch(A + (size_t)(m0 + arow) * lda + k0 + BK + ac0, 0, 1);
            __builtin_prefetch(B + (size_t)(k0 + BK + brow) * ldb + n0 + bc0, 0, 1);
        }
        __syncthreads();

        v16s af[2], bf[4];
#pragma unroll
        for (int mt = 0; mt < 2; mt++)
            af[mt] = frag_ld(&As[wm + mt * 16][0], BK, lane);
#pragma unroll
        for (int nt = 0; nt < 4; nt++)
            bf[nt] = frag_ld(&BsT[wn + nt * 16][0], BK, lane);
#pragma unroll
        for (int mt = 0; mt < 2; mt++)
#pragma unroll
            for (int nt = 0; nt < 4; nt++)
                acc[mt][nt] = wmma_bf16(af[mt], bf[nt], acc[mt][nt]);
    }

#pragma unroll
    for (int mt = 0; mt < 2; mt++)
#pragma unroll
        for (int nt = 0; nt < 4; nt++) {
            int colg = n0 + wn + nt * 16 + mloc;
            float bv = bias ? bias[colg] : 0.0f;
#pragma unroll
            for (int g = 0; g < 8; g++) {
                int rowg = m0 + wm + mt * 16 + g + mhi;
                float v = acc[mt][nt][g] + bv;
                if (relu) v = v > 0.0f ? v : 0.0f;
                if (Cf) Cf[(size_t)rowg * ldc + colg] = v;
                if (Cb) Cb[(size_t)rowg * ldc + colg] = f2bf(v);
            }
        }
}

// ---------------------------------------------------------------- per-head repacks
__global__ void prep_q_kernel(const float* __restrict__ q,
                              const float* __restrict__ rwb, const float* __restrict__ rrb,
                              u16* __restrict__ quO, u16* __restrict__ qvO)
{
    int idx = blockIdx.x * blockDim.x + threadIdx.x;
    if (idx >= BSZ * QLEN * DMODEL) return;
    int d = idx & 63, n = (idx >> 6) & 15, b = (idx >> 10) & 1, i = idx >> 11;
    float v = q[(size_t)(i * 2 + b) * DMODEL + n * 64 + d];
    size_t o = (((size_t)(b * 16 + n)) * QLEN + i) * 64 + d;
    quO[o] = f2bf(v + rwb[n * 64 + d]);
    qvO[o] = f2bf(v + rrb[n * 64 + d]);
}

__global__ void prep_kv_kernel(const float* __restrict__ x, u16* __restrict__ out)
{
    int idx = blockIdx.x * blockDim.x + threadIdx.x;
    if (idx >= BSZ * KLEN * DMODEL) return;
    int d = idx & 63, n = (idx >> 6) & 15, b = (idx >> 10) & 1, j = idx >> 11;
    float v = x[(size_t)(j * 2 + b) * DMODEL + n * 64 + d];
    out[(((size_t)(b * 16 + n)) * KLEN + j) * 64 + d] = f2bf(v);
}

// V stored transposed per head: [b][n][d][j]  (so attention B-fragments are contiguous)
__global__ void prep_vt_kernel(const float* __restrict__ x, u16* __restrict__ out)
{
    int idx = blockIdx.x * blockDim.x + threadIdx.x;
    if (idx >= BSZ * NHEAD * DHEAD * KLEN) return;
    int j = idx & 2047, d = (idx >> 11) & 63, n = (idx >> 17) & 15, b = (idx >> 21) & 1;
    float v = x[(size_t)(j * 2 + b) * DMODEL + n * 64 + d];
    out[idx] = f2bf(v);
}

__global__ void prep_rk_kernel(const float* __restrict__ x, u16* __restrict__ out)
{
    int idx = blockIdx.x * blockDim.x + threadIdx.x;
    if (idx >= KLEN * DMODEL) return;
    int d = idx & 63, n = (idx >> 6) & 15, jr = idx >> 10;
    out[((size_t)n * KLEN + jr) * 64 + d] = f2bf(x[(size_t)jr * DMODEL + n * 64 + d]);
}

// ---------------------------------------------------------------- attention (flash-style)
// grid: (QLEN/64, NHEAD, BSZ), block: 128 threads = 4 waves, wave w owns rows w*16..w*16+15
__global__ __launch_bounds__(128) void attn_kernel(
    const u16* __restrict__ qu_g, const u16* __restrict__ qv_g,
    const u16* __restrict__ k_g,  const u16* __restrict__ vt_g,
    const u16* __restrict__ rk_g, u16* __restrict__ out_bf)
{
    extern __shared__ char smem[];
    u16* qu  = (u16*)smem;        // [64][64]  rows i, cols d
    u16* qv  = qu + 64 * 64;      // [64][64]
    u16* kk  = qv + 64 * 64;      // [64][64]  K tile [j][d]; reused as P tile [i][j]
    u16* vvT = kk + 64 * 64;      // [64][64]  V tile transposed [d][j]
    u16* rkt = vvT + 64 * 64;     // [128][64] r_k band, row-major [jr_local][d]
    float* sS   = (float*)(rkt + 128 * 64); // [64][64] scores
    float* m_s  = sS + 64 * 64;
    float* l_s  = m_s + 64;
    float* al_s = l_s + 64;

    int ib = blockIdx.x, head = blockIdx.y, b = blockIdx.z;
    int i0 = ib * 64;
    int t = threadIdx.x, lane = t & 31, wv = t >> 5;
    int mloc = lane & 15;
    int mhi  = (lane & 16) ? 8 : 0;
    const float scale = 0.125f; // 1/sqrt(64)

    const u16* quB = qu_g + (((size_t)(b * 16 + head)) * QLEN + i0) * 64;
    const u16* qvB = qv_g + (((size_t)(b * 16 + head)) * QLEN + i0) * 64;
    const u16* kB  = k_g  + ((size_t)(b * 16 + head)) * KLEN * 64;
    const u16* vB  = vt_g + ((size_t)(b * 16 + head)) * 64 * KLEN; // [d][j]
    const u16* rkB = rk_g + ((size_t)head) * KLEN * 64;

    { // load q tiles (each 64x64 u16; 128 threads x 32 elems)
        int row = t >> 1, c0 = (t & 1) * 32;
        const uint4* s1 = (const uint4*)(quB + row * 64 + c0);
        uint4* d1 = (uint4*)(qu + row * 64 + c0);
        d1[0] = s1[0]; d1[1] = s1[1]; d1[2] = s1[2]; d1[3] = s1[3];
        const uint4* s2 = (const uint4*)(qvB + row * 64 + c0);
        uint4* d2 = (uint4*)(qv + row * 64 + c0);
        d2[0] = s2[0]; d2[1] = s2[1]; d2[2] = s2[2]; d2[3] = s2[3];
    }
    if (t < 64) { m_s[t] = -3.0e38f; l_s[t] = 0.0f; al_s[t] = 1.0f; }

    v8f oacc[4];
#pragma unroll
    for (int nt = 0; nt < 4; nt++)
#pragma unroll
        for (int g = 0; g < 8; g++) oacc[nt][g] = 0.0f;

    int jbmax = ib + 16; // covers j <= i + MLEN for all rows of this block
    for (int jb = 0; jb <= jbmax; ++jb) {
        int j0 = jb * 64;
        int jr_base = j0 - i0 + 960; // j0 + 1023 - (i0+63)
        __syncthreads();
        { // load K tile [j][d] and V^T tile [d][j]
            int row = t >> 1, c0 = (t & 1) * 32;
            const uint4* sk = (const uint4*)(kB + (size_t)(j0 + row) * 64 + c0);
            uint4* dk = (uint4*)(kk + row * 64 + c0);
            dk[0] = sk[0]; dk[1] = sk[1]; dk[2] = sk[2]; dk[3] = sk[3];
            const uint4* sv = (const uint4*)(vB + (size_t)row * KLEN + j0 + c0);
            uint4* dv = (uint4*)(vvT + row * 64 + c0);
            dv[0] = sv[0]; dv[1] = sv[1]; dv[2] = sv[2]; dv[3] = sv[3];
        }
        { // load r_k band row-major with zero fill: rkt[t][d] = rk[jr_base+t][d]
            int jr = jr_base + t;
            uint4* d = (uint4*)(rkt + t * 64);
            if (jr >= 0 && jr < KLEN) {
                const uint4* s = (const uint4*)(rkB + (size_t)jr * 64);
                d[0] = s[0]; d[1] = s[1]; d[2] = s[2]; d[3] = s[3];
                d[4] = s[4]; d[5] = s[5]; d[6] = s[6]; d[7] = s[7];
            } else {
                uint4 z = {0, 0, 0, 0};
                d[0] = z; d[1] = z; d[2] = z; d[3] = z;
                d[4] = z; d[5] = z; d[6] = z; d[7] = z;
            }
        }
        __syncthreads();

        // ---- AC = (q+u) K^T : wave computes its 16x64 strip
        {
            v8f accA[4];
#pragma unroll
            for (int nt = 0; nt < 4; nt++)
#pragma unroll
                for (int g = 0; g < 8; g++) accA[nt][g] = 0.0f;
#pragma unroll
            for (int ks = 0; ks < 2; ++ks) {
                v16s av = frag_ld(qu + (wv * 16) * 64 + ks * 32, 64, lane);
#pragma unroll
                for (int nt = 0; nt < 4; nt++) {
                    v16s bv = frag_ld(kk + (nt * 16) * 64 + ks * 32, 64, lane);
                    accA[nt] = wmma_bf16(av, bv, accA[nt]);
                }
            }
#pragma unroll
            for (int nt = 0; nt < 4; nt++)
#pragma unroll
                for (int g = 0; g < 8; g++) {
                    int il = wv * 16 + g + mhi;
                    sS[il * 64 + nt * 16 + mloc] = accA[nt][g];
                }
        }
        __syncthreads();

        // ---- BD_pre = (q+v) Rk^T over the jr band; scatter-add with rel-shift
        {
#pragma unroll
            for (int jt = 0; jt < 8; ++jt) {
                v8f accB;
#pragma unroll
                for (int g = 0; g < 8; g++) accB[g] = 0.0f;
#pragma unroll
                for (int ks = 0; ks < 2; ++ks) {
                    v16s av = frag_ld(qv + (wv * 16) * 64 + ks * 32, 64, lane);
                    v16s bv = frag_ld(rkt + (jt * 16) * 64 + ks * 32, 64, lane);
                    accB = wmma_bf16(av, bv, accB);
                }
#pragma unroll
                for (int g = 0; g < 8; g++) {
                    int il = wv * 16 + g + mhi;
                    int jr_l = jt * 16 + mloc;
                    int jl = jr_l + il - 63; // rel-shift: j = jr - 1023 + i
                    if (jl >= 0 && jl < 64) sS[il * 64 + jl] += accB[g];
                }
            }
        }
        __syncthreads();

        // ---- online softmax update; write P (bf16) into kk
        if (t < 64) {
            int r2 = t, ig = i0 + r2;
            int nvalid = ig + MLEN - j0; // j valid while j0+j <= ig+MLEN
            float mold = m_s[r2];
            float mx = mold;
            const float4* srow = (const float4*)(sS + r2 * 64);
#pragma unroll
            for (int q4 = 0; q4 < 16; ++q4) {
                float4 v4 = srow[q4];
                int jbase = q4 * 4;
                if (jbase + 0 <= nvalid) mx = fmaxf(mx, v4.x * scale);
                if (jbase + 1 <= nvalid) mx = fmaxf(mx, v4.y * scale);
                if (jbase + 2 <= nvalid) mx = fmaxf(mx, v4.z * scale);
                if (jbase + 3 <= nvalid) mx = fmaxf(mx, v4.w * scale);
            }
            float alpha = __expf(mold - mx);
            float lsum = l_s[r2] * alpha;
            uint4* prow = (uint4*)(kk + r2 * 64);
#pragma unroll
            for (int q8 = 0; q8 < 8; ++q8) {
                float4 a4 = srow[q8 * 2];
                float4 b4 = srow[q8 * 2 + 1];
                float p[8];
                p[0] = a4.x; p[1] = a4.y; p[2] = a4.z; p[3] = a4.w;
                p[4] = b4.x; p[5] = b4.y; p[6] = b4.z; p[7] = b4.w;
                u32 pk[4];
#pragma unroll
                for (int c = 0; c < 4; ++c) {
                    int j1 = q8 * 8 + 2 * c, j2 = j1 + 1;
                    float p1 = (j1 <= nvalid) ? __expf(p[2 * c] * scale - mx) : 0.0f;
                    float p2 = (j2 <= nvalid) ? __expf(p[2 * c + 1] * scale - mx) : 0.0f;
                    lsum += p1 + p2;
                    pk[c] = (u32)f2bf(p1) | ((u32)f2bf(p2) << 16);
                }
                uint4 pv = {pk[0], pk[1], pk[2], pk[3]};
                prow[q8] = pv;
            }
            m_s[r2] = mx; l_s[r2] = lsum; al_s[r2] = alpha;
        }
        __syncthreads();

        // ---- O = O*alpha + P @ V
        {
#pragma unroll
            for (int nt = 0; nt < 4; nt++)
#pragma unroll
                for (int g = 0; g < 8; g++) {
                    int il = wv * 16 + g + mhi;
                    oacc[nt][g] *= al_s[il];
                }
#pragma unroll
            for (int ks = 0; ks < 2; ++ks) {
                v16s av = frag_ld(kk + (wv * 16) * 64 + ks * 32, 64, lane);
#pragma unroll
                for (int nt = 0; nt < 4; nt++) {
                    v16s bv = frag_ld(vvT + (nt * 16) * 64 + ks * 32, 64, lane);
                    oacc[nt] = wmma_bf16(av, bv, oacc[nt]);
                }
            }
        }
    }
    __syncthreads();

    // write attn_vec[(i*BSZ+b), head*64+d] as bf16
#pragma unroll
    for (int nt = 0; nt < 4; nt++)
#pragma unroll
        for (int g = 0; g < 8; g++) {
            int il = wv * 16 + g + mhi;
            int ig = i0 + il;
            float v = oacc[nt][g] / l_s[il];
            int col = head * 64 + nt * 16 + mloc;
            out_bf[((size_t)ig * BSZ + b) * DMODEL + col] = f2bf(v);
        }
}

// ---------------------------------------------------------------- LayerNorm(X + R)
__global__ __launch_bounds__(256) void ln_kernel(
    const float* __restrict__ X, const float* __restrict__ R,
    const float* __restrict__ gam, const float* __restrict__ bet,
    float* __restrict__ outF, u16* __restrict__ outB)
{
    __shared__ float red[256];
    int row = blockIdx.x, t = threadIdx.x;
    float loc[4];
    float s = 0.0f;
#pragma unroll
    for (int i = 0; i < 4; i++) {
        int c = t + i * 256;
        float v = X[(size_t)row * DMODEL + c] + R[(size_t)row * DMODEL + c];
        loc[i] = v; s += v;
    }
    red[t] = s; __syncthreads();
    for (int st = 128; st > 0; st >>= 1) { if (t < st) red[t] += red[t + st]; __syncthreads(); }
    float mu = red[0] * (1.0f / DMODEL);
    __syncthreads();
    float vs = 0.0f;
#pragma unroll
    for (int i = 0; i < 4; i++) { float d = loc[i] - mu; vs += d * d; }
    red[t] = vs; __syncthreads();
    for (int st = 128; st > 0; st >>= 1) { if (t < st) red[t] += red[t + st]; __syncthreads(); }
    float inv = rsqrtf(red[0] * (1.0f / DMODEL) + 1e-5f);
#pragma unroll
    for (int i = 0; i < 4; i++) {
        int c = t + i * 256;
        float o = (loc[i] - mu) * inv * gam[c] + bet[c];
        if (outF) outF[(size_t)row * DMODEL + c] = o;
        if (outB) outB[(size_t)row * DMODEL + c] = f2bf(o);
    }
}

// ================================================================ host
extern "C" void kernel_launch(void* const* d_in, const int* in_sizes, int n_in,
                              void* d_out, int out_size, void* d_ws, size_t ws_size,
                              hipStream_t stream) {
    (void)in_sizes; (void)n_in; (void)out_size; (void)ws_size;
    const float* w    = (const float*)d_in[0];
    const float* r    = (const float*)d_in[1];
    const float* mems = (const float*)d_in[2];
    const float* Wq = (const float*)d_in[3];  const float* bq = (const float*)d_in[4];
    const float* Wk = (const float*)d_in[5];  const float* bk = (const float*)d_in[6];
    const float* Wv = (const float*)d_in[7];  const float* bv = (const float*)d_in[8];
    const float* Wr = (const float*)d_in[9];  const float* br = (const float*)d_in[10];
    const float* Wo = (const float*)d_in[11]; const float* bo = (const float*)d_in[12];
    const float* rwb = (const float*)d_in[13];
    const float* rrb = (const float*)d_in[14];
    const float* ln1g = (const float*)d_in[15]; const float* ln1b = (const float*)d_in[16];
    const float* W1 = (const float*)d_in[17]; const float* b1 = (const float*)d_in[18];
    const float* W2 = (const float*)d_in[19]; const float* b2 = (const float*)d_in[20];
    const float* ln2g = (const float*)d_in[21]; const float* ln2b = (const float*)d_in[22];

    const int M2 = QLEN * BSZ;  // 2048
    const int M4 = KLEN * BSZ;  // 4096

    char* base = (char*)d_ws; size_t off = 0;
    auto alloc = [&](size_t bytes) { void* p = base + off; off = (off + bytes + 255) & ~(size_t)255; return p; };

    u16* wq_bf = (u16*)alloc((size_t)DMODEL * DMODEL * 2);
    u16* wk_bf = (u16*)alloc((size_t)DMODEL * DMODEL * 2);
    u16* wv_bf = (u16*)alloc((size_t)DMODEL * DMODEL * 2);
    u16* wr_bf = (u16*)alloc((size_t)DMODEL * DMODEL * 2);
    u16* wo_bf = (u16*)alloc((size_t)DMODEL * DMODEL * 2);
    u16* w1_bf = (u16*)alloc((size_t)DMODEL * DINNER * 2);
    u16* w2_bf = (u16*)alloc((size_t)DINNER * DMODEL * 2);
    u16* cat_bf = (u16*)alloc((size_t)M4 * DMODEL * 2);
    u16* r_bf   = (u16*)alloc((size_t)KLEN * DMODEL * 2);
    float* kf32  = (float*)alloc((size_t)M4 * DMODEL * 4);
    float* vf32  = (float*)alloc((size_t)M4 * DMODEL * 4);
    float* qf32  = (float*)alloc((size_t)M2 * DMODEL * 4);
    float* rkf32 = (float*)alloc((size_t)KLEN * DMODEL * 4);
    u16* qu_bf = (u16*)alloc((size_t)BSZ * NHEAD * QLEN * DHEAD * 2);
    u16* qv_bf = (u16*)alloc((size_t)BSZ * NHEAD * QLEN * DHEAD * 2);
    u16* k_bf  = (u16*)alloc((size_t)BSZ * NHEAD * KLEN * DHEAD * 2);
    u16* vt_bf = (u16*)alloc((size_t)BSZ * NHEAD * DHEAD * KLEN * 2);
    u16* rk_bf = (u16*)alloc((size_t)NHEAD * KLEN * DHEAD * 2);
    u16* av_bf = (u16*)alloc((size_t)M2 * DMODEL * 2);
    float* attn_out = (float*)alloc((size_t)M2 * DMODEL * 4);
    float* ln1f = (float*)alloc((size_t)M2 * DMODEL * 4);
    u16*   ln1bf = (u16*)alloc((size_t)M2 * DMODEL * 2);
    u16*   h1bf = (u16*)alloc((size_t)M2 * DINNER * 2);
    float* h2f  = (float*)alloc((size_t)M2 * DMODEL * 4);

    const int T = 256;
    auto blk = [](size_t n, int t) { return (unsigned)((n + t - 1) / t); };

    // weight + activation bf16 conversion
    cvt_bf16_kernel<<<blk((size_t)DMODEL*DMODEL, T), T, 0, stream>>>(Wq, wq_bf, DMODEL*DMODEL);
    cvt_bf16_kernel<<<blk((size_t)DMODEL*DMODEL, T), T, 0, stream>>>(Wk, wk_bf, DMODEL*DMODEL);
    cvt_bf16_kernel<<<blk((size_t)DMODEL*DMODEL, T), T, 0, stream>>>(Wv, wv_bf, DMODEL*DMODEL);
    cvt_bf16_kernel<<<blk((size_t)DMODEL*DMODEL, T), T, 0, stream>>>(Wr, wr_bf, DMODEL*DMODEL);
    cvt_bf16_kernel<<<blk((size_t)DMODEL*DMODEL, T), T, 0, stream>>>(Wo, wo_bf, DMODEL*DMODEL);
    cvt_bf16_kernel<<<blk((size_t)DMODEL*DINNER, T), T, 0, stream>>>(W1, w1_bf, DMODEL*DINNER);
    cvt_bf16_kernel<<<blk((size_t)DINNER*DMODEL, T), T, 0, stream>>>(W2, w2_bf, DINNER*DMODEL);
    // cat = [mems ; w] flattened (row = j*BSZ+b matches flat layouts)
    cvt_bf16_kernel<<<blk((size_t)MLEN*BSZ*DMODEL, T), T, 0, stream>>>(mems, cat_bf, MLEN*BSZ*DMODEL);
    cvt_bf16_kernel<<<blk((size_t)QLEN*BSZ*DMODEL, T), T, 0, stream>>>(w, cat_bf + (size_t)MLEN*BSZ*DMODEL, QLEN*BSZ*DMODEL);
    cvt_bf16_kernel<<<blk((size_t)KLEN*DMODEL, T), T, 0, stream>>>(r, r_bf, KLEN*DMODEL);

    // projections
    gemm_bf16_kernel<<<dim3(DMODEL/BN, M4/BM), 256, 0, stream>>>(cat_bf, DMODEL, wk_bf, DMODEL, bk, kf32, nullptr, DMODEL, M4, DMODEL, DMODEL, 0);
    gemm_bf16_kernel<<<dim3(DMODEL/BN, M4/BM), 256, 0, stream>>>(cat_bf, DMODEL, wv_bf, DMODEL, bv, vf32, nullptr, DMODEL, M4, DMODEL, DMODEL, 0);
    gemm_bf16_kernel<<<dim3(DMODEL/BN, M2/BM), 256, 0, stream>>>(cat_bf + (size_t)MLEN*BSZ*DMODEL, DMODEL, wq_bf, DMODEL, bq, qf32, nullptr, DMODEL, M2, DMODEL, DMODEL, 0);
    gemm_bf16_kernel<<<dim3(DMODEL/BN, KLEN/BM), 256, 0, stream>>>(r_bf, DMODEL, wr_bf, DMODEL, br, rkf32, nullptr, DMODEL, KLEN, DMODEL, DMODEL, 0);

    // per-head packing
    prep_q_kernel<<<blk((size_t)BSZ*QLEN*DMODEL, T), T, 0, stream>>>(qf32, rwb, rrb, qu_bf, qv_bf);
    prep_kv_kernel<<<blk((size_t)BSZ*KLEN*DMODEL, T), T, 0, stream>>>(kf32, k_bf);
    prep_vt_kernel<<<blk((size_t)BSZ*NHEAD*DHEAD*KLEN, T), T, 0, stream>>>(vf32, vt_bf);
    prep_rk_kernel<<<blk((size_t)KLEN*DMODEL, T), T, 0, stream>>>(rkf32, rk_bf);

    // attention
    size_t shm = (size_t)(4 * 64 * 64 + 128 * 64) * sizeof(u16) + (size_t)(64 * 64 + 3 * 64) * sizeof(float);
    attn_kernel<<<dim3(QLEN/64, NHEAD, BSZ), 128, shm, stream>>>(qu_bf, qv_bf, k_bf, vt_bf, rk_bf, av_bf);

    // output projection + ln1
    gemm_bf16_kernel<<<dim3(DMODEL/BN, M2/BM), 256, 0, stream>>>(av_bf, DMODEL, wo_bf, DMODEL, bo, attn_out, nullptr, DMODEL, M2, DMODEL, DMODEL, 0);
    ln_kernel<<<M2, 256, 0, stream>>>(attn_out, w, ln1g, ln1b, ln1f, ln1bf);

    // FFN + ln2 -> d_out
    gemm_bf16_kernel<<<dim3(DINNER/BN, M2/BM), 256, 0, stream>>>(ln1bf, DMODEL, w1_bf, DINNER, b1, nullptr, h1bf, DINNER, M2, DINNER, DMODEL, 1);
    gemm_bf16_kernel<<<dim3(DMODEL/BN, M2/BM), 256, 0, stream>>>(h1bf, DINNER, w2_bf, DMODEL, b2, h2f, nullptr, DMODEL, M2, DMODEL, DINNER, 0);
    ln_kernel<<<M2, 256, 0, stream>>>(h2f, ln1f, ln2g, ln2b, (float*)d_out, nullptr);
}
